// _PointnetSAModuleBase_67508295958600
// MI455X (gfx1250) — matrix operand order repaired and checked by reading, not compiled, not run on hardware
//
#include <hip/hip_runtime.h>
#include <hip/hip_bf16.h>
#include <math.h>

// ---------------- problem constants (from reference) ----------------
#define BB      4
#define NN      8192
#define CIN     64
#define NPOINT  2048
#define NSAMPLE 32
#define RADIUS  0.2f
#define C0      67     // 3 + CIN
#define C1      64
#define C2      128
#define CXYZ    10

// padded f16 weight strides (K multiples of 32, even -> dword-aligned frags)
#define K1P     96     // 67 -> 96
#define K2P     64
#define KAP     160    // 138 -> 160
#define KFP     128

// fused tile: 4 centers * 32 samples = 128 points per workgroup
#define TILE_S  4
#define PTS     (TILE_S * NSAMPLE)   // 128

typedef __attribute__((ext_vector_type(16))) _Float16 v16h;
typedef __attribute__((ext_vector_type(8)))  float    v8f;

union Frag16 { v16h h; unsigned u[8]; };

// ---- A-fragment (16x32 f16, M x K) from row-major f16 weights in global ----
// lane l holds row M = l&15; VGPR v holds halves K = kb, kb+1 where
// kb = kbase + 2v + (v>=4 ? 8 : 0) + (l>=16 ? 8 : 0)        [ISA 7.12.2]
__device__ __forceinline__ v16h load_a_frag(const _Float16* __restrict__ W,
                                            int stride, int mbase, int kbase, int lane) {
  Frag16 f;
  const int row = mbase + (lane & 15);
  const int khi = (lane >= 16) ? 8 : 0;
  const _Float16* rp = W + row * stride;
#pragma unroll
  for (int v = 0; v < 8; ++v) {
    int k = kbase + 2 * v + ((v >= 4) ? 8 : 0) + khi;
    f.u[v] = *(const unsigned*)(rp + k);
  }
  return f.h;
}

// ---- B-fragment (32x16 f16, K x N) from point-major LDS [point][stride] ----
// lane l holds column N = l&15; VGPR v holds K = kbase + 2v + (v>=4 ? 8 : 0) + (l>=16 ? 8 : 0)
__device__ __forceinline__ v16h load_b_frag(const _Float16* Fs,
                                            int stride, int nbase, int kbase, int lane) {
  Frag16 f;
  const int p   = nbase + (lane & 15);
  const int khi = (lane >= 16) ? 8 : 0;
  const _Float16* rp = Fs + p * stride;
#pragma unroll
  for (int v = 0; v < 8; ++v) {
    int k = kbase + 2 * v + ((v >= 4) ? 8 : 0) + khi;
    f.u[v] = *(const unsigned*)(rp + k);
  }
  return f.h;
}

#define WMMA_F16(A, Bm, C) \
  __builtin_amdgcn_wmma_f32_16x16x32_f16(false, (A), false, (Bm), (short)0, (C), false, false)

// =====================================================================
// Kernel 1: furthest point sampling, one workgroup per batch.
// Points live in registers (8 per thread, strided p = t + i*1024).
// =====================================================================
__global__ __launch_bounds__(1024) void fps_kernel(const float* __restrict__ xyz,
                                                   int* __restrict__ fps_idx) {
  __shared__ float rv[32];
  __shared__ int   ri[32];
  __shared__ float cen_s[3];
  __shared__ int   far_s;

  const int b = blockIdx.x;
  const int t = threadIdx.x;
  const float* X = xyz + (size_t)b * NN * 3;

  float px[8], py[8], pz[8], dd[8];
#pragma unroll
  for (int i = 0; i < 8; ++i) {
    int p = t + i * 1024;
    px[i] = X[p * 3 + 0];
    py[i] = X[p * 3 + 1];
    pz[i] = X[p * 3 + 2];
    dd[i] = 1e10f;
  }

  int far = 0;
  for (int it = 0; it < NPOINT; ++it) {
    if (t == (far & 1023)) {
      int sl = far >> 10;
      cen_s[0] = px[sl]; cen_s[1] = py[sl]; cen_s[2] = pz[sl];
    }
    if (t == 0) fps_idx[b * NPOINT + it] = far;   // reference records incoming 'far'
    __syncthreads();

    const float cx = cen_s[0], cy = cen_s[1], cz = cen_s[2];
    float bv = -1.0f; int bi = 0;
#pragma unroll
    for (int i = 0; i < 8; ++i) {
      float dx = px[i] - cx, dy = py[i] - cy, dz = pz[i] - cz;
      float d = dx * dx + dy * dy + dz * dz;
      dd[i] = fminf(dd[i], d);
      int p = t + i * 1024;
      if (dd[i] > bv || (dd[i] == bv && p < bi)) { bv = dd[i]; bi = p; }
    }
    // wave32 reduce (argmax, lowest index on tie, matching jnp.argmax)
#pragma unroll
    for (int off = 16; off > 0; off >>= 1) {
      float ov = __shfl_down(bv, off, 32);
      int   oi = __shfl_down(bi, off, 32);
      if (ov > bv || (ov == bv && oi < bi)) { bv = ov; bi = oi; }
    }
    if ((t & 31) == 0) { rv[t >> 5] = bv; ri[t >> 5] = bi; }
    __syncthreads();
    if (t < 32) {
      bv = rv[t]; bi = ri[t];
#pragma unroll
      for (int off = 16; off > 0; off >>= 1) {
        float ov = __shfl_down(bv, off, 32);
        int   oi = __shfl_down(bi, off, 32);
        if (ov > bv || (ov == bv && oi < bi)) { bv = ov; bi = oi; }
      }
      if (t == 0) far_s = bi;
    }
    __syncthreads();
    far = far_s;
  }
}

// =====================================================================
// Kernel 2: ball query (one wave32 per center) + new_xyz output.
// First NSAMPLE in-radius points in index order (torch semantics).
// =====================================================================
__global__ __launch_bounds__(256) void ball_kernel(const float* __restrict__ xyz,
                                                   const int* __restrict__ fps_idx,
                                                   int* __restrict__ ball_idx,
                                                   float* __restrict__ new_xyz) {
  const int w    = blockIdx.x * (blockDim.x >> 5) + (threadIdx.x >> 5);
  const int lane = threadIdx.x & 31;
  const int b = w / NPOINT, s = w % NPOINT;
  const float* X = xyz + (size_t)b * NN * 3;

  const int ci = fps_idx[b * NPOINT + s];
  const float cx = X[ci * 3 + 0], cy = X[ci * 3 + 1], cz = X[ci * 3 + 2];
  if (lane == 0) {
    float* nx = new_xyz + (size_t)(b * NPOINT + s) * 3;
    nx[0] = cx; nx[1] = cy; nx[2] = cz;
  }
  int* outp = ball_idx + (size_t)(b * NPOINT + s) * NSAMPLE;
  const float r2 = RADIUS * RADIUS;

  int cnt = 0, first = -1;
  for (int base = 0; base < NN && cnt < NSAMPLE; base += 32) {
    int p = base + lane;
    float dx = X[p * 3 + 0] - cx, dy = X[p * 3 + 1] - cy, dz = X[p * 3 + 2] - cz;
    bool ok = (dx * dx + dy * dy + dz * dz) < r2;
    unsigned mm = (unsigned)__ballot(ok);
    if (first < 0 && mm) first = base + (__ffs(mm) - 1);
    int pos = cnt + __popc(mm & ((1u << lane) - 1u));
    if (ok && pos < NSAMPLE) outp[pos] = p;
    cnt += __popc(mm);
  }
  if (cnt < NSAMPLE) {
    for (int q = cnt + lane; q < NSAMPLE; q += 32) outp[q] = first;
  }
}

// =====================================================================
// Kernel 3: convert/pad weights to f16 with WMMA-friendly strides.
// =====================================================================
__global__ void prep_weights(const float* __restrict__ W1, const float* __restrict__ W2,
                             const float* __restrict__ Wa, const float* __restrict__ Wf,
                             _Float16* __restrict__ W1h, _Float16* __restrict__ W2h,
                             _Float16* __restrict__ Wah, _Float16* __restrict__ Wfh) {
  int i = blockIdx.x * blockDim.x + threadIdx.x;
  if (i < C1 * K1P) {                       // W1: 64 x 67 -> 64 x 96
    int o = i / K1P, k = i % K1P;
    W1h[i] = (_Float16)(k < C0 ? W1[o * C0 + k] : 0.0f);
    return;
  }
  i -= C1 * K1P;
  if (i < C2 * K2P) { W2h[i] = (_Float16)W2[i]; return; }   // 128 x 64
  i -= C2 * K2P;
  if (i < C2 * KAP) {                       // Wa: 128 x 138 -> 128 x 160
    int o = i / KAP, k = i % KAP;
    Wah[i] = (_Float16)(k < (C2 + CXYZ) ? Wa[o * (C2 + CXYZ) + k] : 0.0f);
    return;
  }
  i -= C2 * KAP;
  if (i < C2 * KFP) { Wfh[i] = (_Float16)Wf[i]; }           // 128 x 128
}

// =====================================================================
// Kernel 4: fused gather + MLP1 + MLP2 + attention + softmax-pool + Wf.
// One workgroup (256 threads = 8 waves) per 4 centers (128 points).
// All intermediates in LDS; all GEMMs via v_wmma_f32_16x16x32_f16.
// =====================================================================
__global__ __launch_bounds__(256) void fused_kernel(
    const float* __restrict__ xyz, const float* __restrict__ feats,
    const int* __restrict__ ball_idx,
    const _Float16* __restrict__ W1h, const float* __restrict__ b1,
    const _Float16* __restrict__ W2h, const float* __restrict__ b2,
    const _Float16* __restrict__ Wah, const float* __restrict__ ba,
    const _Float16* __restrict__ Wfh, const float* __restrict__ bf,
    const float* __restrict__ new_xyz, float* __restrict__ out) {
  extern __shared__ char smem[];
  _Float16* feat_s = (_Float16*)smem;               // [128][96]  f16
  _Float16* xyzs_s = feat_s + PTS * K1P;            // [128][32]  f16
  _Float16* h1_s   = xyzs_s + PTS * 32;             // [128][64]  f16
  _Float16* h2_s   = h1_s   + PTS * C1;             // [128][128] f16
  float*    s_s    = (float*)(h2_s + PTS * C2);     // [16][128]  f32 score chunk
  _Float16* pool_s = (_Float16*)(s_s + 16 * PTS);   // [16][128]  f16 (4 valid cols)

  const int g  = blockIdx.x;
  const int b  = g / (NPOINT / TILE_S);
  const int s0 = (g % (NPOINT / TILE_S)) * TILE_S;
  const int t  = threadIdx.x;
  const int wave = t >> 5, lane = t & 31;

  // ---- zero padded activation regions (feat+xyzs contiguous, and pool) ----
  {
    unsigned* z1 = (unsigned*)feat_s;                   // (128*96 + 128*32)*2 /4
    const int n1 = (PTS * K1P + PTS * 32) >> 1;
    for (int i = t; i < n1; i += 256) z1[i] = 0u;
    unsigned* z2 = (unsigned*)pool_s;                   // 16*128*2 /4
    const int n2 = (16 * C2) >> 1;
    for (int i = t; i < n2; i += 256) z2[i] = 0u;
  }
  __syncthreads();

  // ---- gather: rel/dist/cen/nbr + 64 feature channels per point ----
  {
    const int p = t & 127;          // point in tile
    const int hs = t >> 7;          // 0/1 -> channels 0..31 / 32..63
    const int sc = p >> 5;          // center within tile
    const int kk = p & 31;
    const int idx = ball_idx[(size_t)((b * NPOINT + s0 + sc) * NSAMPLE) + kk];
    const float nx = xyz[((size_t)b * NN + idx) * 3 + 0];
    const float ny = xyz[((size_t)b * NN + idx) * 3 + 1];
    const float nz = xyz[((size_t)b * NN + idx) * 3 + 2];
    const float* cp = new_xyz + (size_t)(b * NPOINT + s0 + sc) * 3;
    const float cx = cp[0], cy = cp[1], cz = cp[2];
    const float rx = nx - cx, ry = ny - cy, rz = nz - cz;
    if (hs == 0) {
      _Float16* fp = feat_s + p * K1P;
      fp[0] = (_Float16)rx; fp[1] = (_Float16)ry; fp[2] = (_Float16)rz;
      _Float16* xp = xyzs_s + p * 32;
      xp[0] = (_Float16)sqrtf(rx * rx + ry * ry + rz * rz);
      xp[1] = (_Float16)rx; xp[2] = (_Float16)ry; xp[3] = (_Float16)rz;
      xp[4] = (_Float16)cx; xp[5] = (_Float16)cy; xp[6] = (_Float16)cz;
      xp[7] = (_Float16)nx; xp[8] = (_Float16)ny; xp[9] = (_Float16)nz;
    }
    const int cb = hs * 32;
    const float* fbase = feats + ((size_t)b * CIN + cb) * NN + idx;
#pragma unroll 4
    for (int c = 0; c < 32; ++c)
      feat_s[p * K1P + 3 + cb + c] = (_Float16)fbase[(size_t)c * NN];
  }
  __syncthreads();

  // ---- layer 1: H1 = relu(W1 * feat + b1), M=64, N=128, K=96 ----
  for (int j = 0; j < 4; ++j) {
    const int job = wave * 4 + j;           // 32 tile jobs
    const int mt = job >> 3, nt = job & 7;
    v8f acc = {};
#pragma unroll
    for (int kt = 0; kt < 3; ++kt) {
      v16h a  = load_a_frag(W1h, K1P, mt * 16, kt * 32, lane);
      v16h bm = load_b_frag(feat_s, K1P, nt * 16, kt * 32, lane);
      acc = WMMA_F16(a, bm, acc);
    }
    const int n = nt * 16 + (lane & 15);
    const int mofs = (lane >= 16) ? 8 : 0;
#pragma unroll
    for (int r = 0; r < 8; ++r) {
      int m = mt * 16 + r + mofs;
      float v = acc[r] + b1[m];
      h1_s[n * C1 + m] = (_Float16)(v > 0.0f ? v : 0.0f);
    }
  }
  __syncthreads();

  // ---- layer 2: H2 = relu(W2 * H1 + b2), M=128, N=128, K=64 ----
  for (int j = 0; j < 8; ++j) {
    const int job = wave * 8 + j;           // 64 tile jobs
    const int mt = job >> 3, nt = job & 7;
    v8f acc = {};
#pragma unroll
    for (int kt = 0; kt < 2; ++kt) {
      v16h a  = load_a_frag(W2h, K2P, mt * 16, kt * 32, lane);
      v16h bm = load_b_frag(h1_s, C1, nt * 16, kt * 32, lane);
      acc = WMMA_F16(a, bm, acc);
    }
    const int n = nt * 16 + (lane & 15);
    const int mofs = (lane >= 16) ? 8 : 0;
#pragma unroll
    for (int r = 0; r < 8; ++r) {
      int m = mt * 16 + r + mofs;
      float v = acc[r] + b2[m];
      h2_s[n * C2 + m] = (_Float16)(v > 0.0f ? v : 0.0f);
    }
  }
  __syncthreads();

  // ---- attention: S = Wa * [H2; xyz_set] + ba, chunked 16 out-chans ----
  for (int chunk = 0; chunk < 8; ++chunk) {
    {
      const int nt = wave;                  // 8 column tiles, 1 per wave
      v8f acc = {};
#pragma unroll
      for (int kt = 0; kt < 4; ++kt) {      // feat part, K=128
        v16h a  = load_a_frag(Wah, KAP, chunk * 16, kt * 32, lane);
        v16h bm = load_b_frag(h2_s, C2, nt * 16, kt * 32, lane);
        acc = WMMA_F16(a, bm, acc);
      }
      {                                     // xyz part, K=32 (10 real + pad)
        v16h a  = load_a_frag(Wah, KAP, chunk * 16, 128, lane);
        v16h bm = load_b_frag(xyzs_s, 32, nt * 16, 0, lane);
        acc = WMMA_F16(a, bm, acc);
      }
      const int n = nt * 16 + (lane & 15);
      const int mofs = (lane >= 16) ? 8 : 0;
#pragma unroll
      for (int r = 0; r < 8; ++r) {
        int mloc = r + mofs;
        s_s[mloc * PTS + n] = acc[r] + ba[chunk * 16 + mloc];
      }
    }
    __syncthreads();
    // softmax over k=32 per (out-chan, center), then attention pooling
    if (t < 64) {
      const int oloc = t >> 2, sc = t & 3;
      const float* row = s_s + oloc * PTS + sc * 32;
      float mx = -1e30f;
#pragma unroll 8
      for (int k = 0; k < 32; ++k) mx = fmaxf(mx, row[k]);
      const int o = chunk * 16 + oloc;
      float sum = 0.0f, pool = 0.0f;
#pragma unroll 8
      for (int k = 0; k < 32; ++k) {
        float e = __expf(row[k] - mx);
        sum += e;
        pool += e * (float)h2_s[(sc * 32 + k) * C2 + o];
      }
      pool_s[sc * C2 + o] = (_Float16)(pool / sum);
    }
    __syncthreads();
  }

  // ---- final: out = Wf * pooled + bf, M=128, N=16(pad of 4), K=128 ----
  {
    const int mt = wave;                    // 8 row tiles, 1 per wave
    v8f acc = {};
#pragma unroll
    for (int kt = 0; kt < 4; ++kt) {
      v16h a  = load_a_frag(Wfh, KFP, mt * 16, kt * 32, lane);
      v16h bm = load_b_frag(pool_s, C2, 0, kt * 32, lane);
      acc = WMMA_F16(a, bm, acc);
    }
    const int n = lane & 15;
    const int mofs = (lane >= 16) ? 8 : 0;
    if (n < TILE_S) {
#pragma unroll
      for (int r = 0; r < 8; ++r) {
        int m = mt * 16 + r + mofs;
        out[((size_t)b * C2 + m) * NPOINT + s0 + n] = acc[r] + bf[m];
      }
    }
  }
}

// =====================================================================
extern "C" void kernel_launch(void* const* d_in, const int* in_sizes, int n_in,
                              void* d_out, int out_size, void* d_ws, size_t ws_size,
                              hipStream_t stream) {
  const float* xyz   = (const float*)d_in[0];
  const float* feats = (const float*)d_in[1];
  const float* W1    = (const float*)d_in[2];
  const float* b1    = (const float*)d_in[3];
  const float* W2    = (const float*)d_in[4];
  const float* b2    = (const float*)d_in[5];
  const float* Wa    = (const float*)d_in[6];
  const float* ba    = (const float*)d_in[7];
  const float* Wf    = (const float*)d_in[8];
  const float* bf    = (const float*)d_in[9];

  float* out_all  = (float*)d_out;
  float* new_xyz  = out_all;                         // [B][NPOINT][3]
  float* out_feat = out_all + (size_t)BB * NPOINT * 3;  // [B][C2][NPOINT]

  char* ws = (char*)d_ws;
  size_t off = 0;
  int* fps_idx  = (int*)(ws + off); off += (size_t)BB * NPOINT * 4;
  int* ball_idx = (int*)(ws + off); off += (size_t)BB * NPOINT * NSAMPLE * 4;
  _Float16* W1h = (_Float16*)(ws + off); off += (size_t)C1 * K1P * 2;
  _Float16* W2h = (_Float16*)(ws + off); off += (size_t)C2 * K2P * 2;
  _Float16* Wah = (_Float16*)(ws + off); off += (size_t)C2 * KAP * 2;
  _Float16* Wfh = (_Float16*)(ws + off); off += (size_t)C2 * KFP * 2;

  fps_kernel<<<BB, 1024, 0, stream>>>(xyz, fps_idx);
  ball_kernel<<<(BB * NPOINT) / 8, 256, 0, stream>>>(xyz, fps_idx, ball_idx, new_xyz);

  const int wtotal = C1 * K1P + C2 * K2P + C2 * KAP + C2 * KFP;   // 51200
  prep_weights<<<(wtotal + 255) / 256, 256, 0, stream>>>(W1, W2, Wa, Wf, W1h, W2h, Wah, Wfh);

  const size_t lds_bytes =
      (size_t)PTS * K1P * 2 +   // feat_s
      (size_t)PTS * 32 * 2 +    // xyzs_s
      (size_t)PTS * C1 * 2 +    // h1_s
      (size_t)PTS * C2 * 2 +    // h2_s
      (size_t)16 * PTS * 4 +    // s_s
      (size_t)16 * C2 * 2;      // pool_s      (= 94208 bytes)

  fused_kernel<<<(BB * NPOINT) / TILE_S, 256, lds_bytes, stream>>>(
      xyz, feats, ball_idx, W1h, b1, W2h, b2, Wah, ba, Wfh, bf, new_xyz, out_feat);
}
